// PPGPeakPerformer_68118181315186
// MI455X (gfx1250) — matrix operand order repaired and checked by reading, not compiled, not run on hardware
//
#include <hip/hip_runtime.h>
#include <hip/hip_bf16.h>

typedef __bf16 bf16;
typedef __attribute__((ext_vector_type(16))) __bf16 v16bf;
typedef __attribute__((ext_vector_type(8)))  __bf16 v8bf;
typedef __attribute__((ext_vector_type(8)))  float  v8f;

#define SL 4096
#define NL 4

__device__ __forceinline__ float geluf(float x) {
    return 0.5f * x * (1.0f + erff(x * 0.7071067811865475f));
}
__device__ __forceinline__ float softplusf(float x) {
    return (x > 20.f) ? x : log1pf(expf(x));
}

// ---------------------------------------------------------------------------
// Weight packing: f32 [K][N] -> bf16 [N][K] (transposed, WMMA-B friendly)
// ---------------------------------------------------------------------------
__global__ void pack_t(const float* __restrict__ src, bf16* __restrict__ dst,
                       int K, int N) {
    int i = blockIdx.x * 256 + threadIdx.x;
    if (i >= K * N) return;
    int n = i / K, k = i - n * K;
    dst[i] = (bf16)src[(size_t)k * N + n];
}

__global__ void pack_bias(const float* __restrict__ bq, const float* __restrict__ bk,
                          const float* __restrict__ bv, float* __restrict__ dst) {
    int i = blockIdx.x * 256 + threadIdx.x;
    if (i >= NL * 768) return;
    int l = i / 768, j = i - l * 768;
    float v = (j < 256) ? bq[l * 256 + j]
            : (j < 512) ? bk[l * 256 + j - 256]
                        : bv[l * 256 + j - 512];
    dst[i] = v;
}

// ---------------------------------------------------------------------------
// WMMA GEMM: C = epi(A[M,K] @ Bt[N,K]^T + bias)
//   Block tile 128 x BN, 8 waves as WM x (8/WM); each wave owns
//   (128/WM) x (BN*WM/8) and issues MF*NF WMMAs per 32-wide K-step with all
//   fragments preloaded (back-to-back v_wmma issue). LDS double-buffered:
//   K-step k+1 fetched from global into regs while buffer k feeds WMMAs.
//   EPI: 0 = bf16 store, 1 = GELU->bf16, 2 = f32 residual add (C += v),
//        3 = softplus->bf16
//   Batched over blockIdx.z with (b = z/zH, h = z%zH) strides.
// ---------------------------------------------------------------------------
template <int EPI, int BN, int WM>
__global__ __launch_bounds__(256) void gemm_wmma(
    const bf16* __restrict__ A, size_t lda,
    const bf16* __restrict__ Bt,
    const float* __restrict__ bias,
    void* __restrict__ Cv, size_t ldc,
    int M, int N, int K,
    int zH, size_t sAb, size_t sAh, size_t sBb, size_t sBh,
    size_t sCb, size_t sCh)
{
    constexpr int WN  = 8 / WM;            // waves in N
    constexpr int MF  = 128 / (WM * 16);   // M fragments per wave
    constexpr int NF  = BN / (WN * 16);    // N fragments per wave
    constexpr int BCH = BN / 64;           // B v8bf chunks per thread

    __shared__ bf16 lA[2][128 * 48];       // stride 48 elems (96B, 16B aligned)
    __shared__ bf16 lB[2][BN * 48];

    const int t = threadIdx.x;
    const int w = t >> 5, lane = t & 31;
    const int half = lane >> 4, l16 = lane & 15;
    const int wm = w % WM, wn = w / WM;

    const int z = blockIdx.z;
    const int zb = z / zH, zh = z - zb * zH;
    A  += (size_t)zb * sAb + (size_t)zh * sAh;
    Bt += (size_t)zb * sBb + (size_t)zh * sBh;
    const size_t cofs = (size_t)zb * sCb + (size_t)zh * sCh;

    const int bm = blockIdx.y * 128, bn = blockIdx.x * BN;

    v8f acc[MF][NF];
    const v8f vzero = {0.f, 0.f, 0.f, 0.f, 0.f, 0.f, 0.f, 0.f};
#pragma unroll
    for (int i = 0; i < MF; ++i)
#pragma unroll
        for (int j = 0; j < NF; ++j) acc[i][j] = vzero;

    // per-thread staging coordinates (tile row / 8-elem column chunk)
    int arow[2], acol[2], brow[BCH], bcol[BCH];
#pragma unroll
    for (int c = 0; c < 2; ++c) {
        int id = t + c * 256;
        arow[c] = id >> 2;
        acol[c] = (id & 3) * 8;
    }
#pragma unroll
    for (int c = 0; c < BCH; ++c) {
        int id = t + c * 256;
        brow[c] = id >> 2;
        bcol[c] = (id & 3) * 8;
    }

    const int nk = K >> 5;
    v8bf areg[2], breg[BCH];

    // prologue: fetch K-step 0 and stage into buffer 0
#pragma unroll
    for (int c = 0; c < 2; ++c)
        areg[c] = *(const v8bf*)(A + (size_t)(bm + arow[c]) * lda + acol[c]);
#pragma unroll
    for (int c = 0; c < BCH; ++c)
        breg[c] = *(const v8bf*)(Bt + (size_t)(bn + brow[c]) * K + bcol[c]);
#pragma unroll
    for (int c = 0; c < 2; ++c)
        *(v8bf*)&lA[0][arow[c] * 48 + acol[c]] = areg[c];
#pragma unroll
    for (int c = 0; c < BCH; ++c)
        *(v8bf*)&lB[0][brow[c] * 48 + bcol[c]] = breg[c];
    __syncthreads();

    for (int kt = 0; kt < nk; ++kt) {
        const int cur = kt & 1;
        // fetch next K-step from global while WMMAs run on current buffer
        if (kt + 1 < nk) {
            const int ko = (kt + 1) * 32;
#pragma unroll
            for (int c = 0; c < 2; ++c)
                areg[c] = *(const v8bf*)(A + (size_t)(bm + arow[c]) * lda + ko + acol[c]);
#pragma unroll
            for (int c = 0; c < BCH; ++c)
                breg[c] = *(const v8bf*)(Bt + (size_t)(bn + brow[c]) * K + ko + bcol[c]);
        }

        // preload ALL fragments (distinct regs), then issue WMMAs back-to-back
        v16bf av[MF], bv[NF];
#pragma unroll
        for (int mf = 0; mf < MF; ++mf) {
            const bf16* pa = &lA[cur][(wm * (MF * 16) + mf * 16 + l16) * 48 + half * 8];
            v8bf a0 = *(const v8bf*)pa;
            v8bf a1 = *(const v8bf*)(pa + 16);
            av[mf] = __builtin_shufflevector(a0, a1, 0, 1, 2, 3, 4, 5, 6, 7,
                                             8, 9, 10, 11, 12, 13, 14, 15);
        }
#pragma unroll
        for (int nf = 0; nf < NF; ++nf) {
            const bf16* pb = &lB[cur][(wn * (NF * 16) + nf * 16 + l16) * 48 + half * 8];
            v8bf b0 = *(const v8bf*)pb;
            v8bf b1 = *(const v8bf*)(pb + 16);
            bv[nf] = __builtin_shufflevector(b0, b1, 0, 1, 2, 3, 4, 5, 6, 7,
                                             8, 9, 10, 11, 12, 13, 14, 15);
        }
#pragma unroll
        for (int mf = 0; mf < MF; ++mf)
#pragma unroll
            for (int nf = 0; nf < NF; ++nf)
                acc[mf][nf] = __builtin_amdgcn_wmma_f32_16x16x32_bf16(
                    false, av[mf], false, bv[nf], (short)0, acc[mf][nf], false, false);

        // stage next K-step into the other buffer
        if (kt + 1 < nk) {
            const int nxt = cur ^ 1;
#pragma unroll
            for (int c = 0; c < 2; ++c)
                *(v8bf*)&lA[nxt][arow[c] * 48 + acol[c]] = areg[c];
#pragma unroll
            for (int c = 0; c < BCH; ++c)
                *(v8bf*)&lB[nxt][brow[c] * 48 + bcol[c]] = breg[c];
        }
        __syncthreads();
    }

#pragma unroll
    for (int mf = 0; mf < MF; ++mf) {
#pragma unroll
        for (int nf = 0; nf < NF; ++nf) {
            int col = bn + wn * (NF * 16) + nf * 16 + l16;
            if (col >= N) continue;
            float bsv = bias ? bias[col] : 0.f;
#pragma unroll
            for (int r = 0; r < 8; ++r) {
                int row = bm + wm * (MF * 16) + mf * 16 + half * 8 + r;
                if (row >= M) continue;
                float v = acc[mf][nf][r] + bsv;
                size_t idx = cofs + (size_t)row * ldc + col;
                if (EPI == 0)      ((bf16*)Cv)[idx] = (bf16)v;
                else if (EPI == 1) ((bf16*)Cv)[idx] = (bf16)geluf(v);
                else if (EPI == 3) ((bf16*)Cv)[idx] = (bf16)softplusf(v);
                else               ((float*)Cv)[idx] += v;  // residual into h
            }
        }
    }
}

// ---------------------------------------------------------------------------
// LayerNorm rows of 256 f32 -> scaled/shifted bf16 (one wave per row)
// ---------------------------------------------------------------------------
__global__ __launch_bounds__(256) void ln_rows(
    const float* __restrict__ X, const float* __restrict__ g,
    const float* __restrict__ b, bf16* __restrict__ O, int nrows)
{
    int w = threadIdx.x >> 5, lane = threadIdx.x & 31;
    int row = blockIdx.x * 8 + w;
    if (row >= nrows) return;
    const float* xr = X + (size_t)row * 256 + lane * 8;
    float4 u0 = *(const float4*)xr;
    float4 u1 = *(const float4*)(xr + 4);
    float v[8] = {u0.x, u0.y, u0.z, u0.w, u1.x, u1.y, u1.z, u1.w};
    float s = 0.f, q = 0.f;
#pragma unroll
    for (int j = 0; j < 8; ++j) { s += v[j]; q += v[j] * v[j]; }
    for (int o = 16; o > 0; o >>= 1) {
        s += __shfl_xor(s, o);
        q += __shfl_xor(q, o);
    }
    float mean = s * (1.f / 256.f);
    float var  = q * (1.f / 256.f) - mean * mean;
    float rstd = rsqrtf(var + 1e-5f);
    v8bf o8;
#pragma unroll
    for (int j = 0; j < 8; ++j) {
        int col = lane * 8 + j;
        o8[j] = (bf16)((v[j] - mean) * rstd * g[col] + b[col]);
    }
    *(v8bf*)(O + (size_t)row * 256 + lane * 8) = o8;
}

// ---------------------------------------------------------------------------
// Performer: partial kv[f][d] = sum_s kp[s,f]*v[s,d], ksum[f] = sum_s kp[s,f]
// grid (8 splits of 512 s, B_c*H). No atomics (graph-safe).
// ---------------------------------------------------------------------------
__global__ __launch_bounds__(256) void kv_partial(
    const bf16* __restrict__ qkv, const bf16* __restrict__ kp,
    float* __restrict__ pkv, float* __restrict__ pks)
{
    const int z = blockIdx.y, sp = blockIdx.x;
    const int b = z >> 3, hh = z & 7;
    const bf16* V  = qkv + (size_t)b * SL * 768 + 512 + hh * 32;  // v slice
    const bf16* KP = kp + (size_t)z * SL * 64;
    const int t = threadIdx.x;
    const int d = t & 31, f0 = (t >> 5) * 8;
    float acc[8] = {0, 0, 0, 0, 0, 0, 0, 0};
    float ks[8]  = {0, 0, 0, 0, 0, 0, 0, 0};
    const int s0 = sp * 512;
    for (int s = s0; s < s0 + 512; ++s) {
        float vs = (float)V[(size_t)s * 768 + d];
        v8bf kk = *(const v8bf*)(KP + (size_t)s * 64 + f0);
#pragma unroll
        for (int j = 0; j < 8; ++j) {
            float kj = (float)kk[j];
            acc[j] += kj * vs;
            ks[j]  += kj;
        }
    }
    float* po = pkv + ((size_t)z * 8 + sp) * 2048;
#pragma unroll
    for (int j = 0; j < 8; ++j) po[(f0 + j) * 32 + d] = acc[j];
    if (d == 0) {
        float* ps = pks + ((size_t)z * 8 + sp) * 64;
#pragma unroll
        for (int j = 0; j < 8; ++j) ps[f0 + j] = ks[j];
    }
}

// ---------------------------------------------------------------------------
// Performer combine: out = (qp @ kv) / (qp @ ksum + 1e-8), heads merged.
// Folds the split-S partial reduction into LDS staging.
// ---------------------------------------------------------------------------
__global__ __launch_bounds__(256) void attn_combine(
    const bf16* __restrict__ qp, const float* __restrict__ pkv,
    const float* __restrict__ pks, bf16* __restrict__ attn)
{
    __shared__ float kvs[2048];
    __shared__ float ksl[64];
    const int t = threadIdx.x;
    const int z = blockIdx.y;
    const int b = z >> 3, hh = z & 7;
    for (int i = t; i < 2048; i += 256) {
        float s = 0.f;
        const float* p = pkv + (size_t)z * 8 * 2048 + i;
#pragma unroll
        for (int sp = 0; sp < 8; ++sp) s += p[(size_t)sp * 2048];
        kvs[i] = s;
    }
    if (t < 64) {
        float s = 0.f;
        const float* p = pks + (size_t)z * 8 * 64 + t;
#pragma unroll
        for (int sp = 0; sp < 8; ++sp) s += p[(size_t)sp * 64];
        ksl[t] = s;
    }
    __syncthreads();

    const int s = blockIdx.x * 256 + t;
    const bf16* qr = qp + ((size_t)z * SL + s) * 64;
    float o[32];
#pragma unroll
    for (int d = 0; d < 32; ++d) o[d] = 0.f;
    float den = 1e-8f;
    for (int g = 0; g < 8; ++g) {
        v8bf q8 = *(const v8bf*)(qr + g * 8);
#pragma unroll
        for (int j = 0; j < 8; ++j) {
            float qf = (float)q8[j];
            int f = g * 8 + j;
            den += qf * ksl[f];
#pragma unroll
            for (int d = 0; d < 32; ++d) o[d] += qf * kvs[f * 32 + d];
        }
    }
    float inv = 1.f / den;
    bf16* op = attn + ((size_t)b * SL + s) * 256 + hh * 32;
#pragma unroll
    for (int g = 0; g < 4; ++g) {
        v8bf o8;
#pragma unroll
        for (int j = 0; j < 8; ++j) o8[j] = (bf16)(o[g * 8 + j] * inv);
        *(v8bf*)(op + g * 8) = o8;
    }
}

// ---------------------------------------------------------------------------
// Embedding: conv(1->128,k7,p3)+BN+GELU, conv(128->256,k5,p2)+BN+GELU, +posenc
// ---------------------------------------------------------------------------
__global__ __launch_bounds__(256) void embed_conv(
    const float* __restrict__ x,
    const float* __restrict__ w1, const float* __restrict__ b1,
    const float* __restrict__ g1, const float* __restrict__ bb1,
    const float* __restrict__ w2, const float* __restrict__ b2,
    const float* __restrict__ g2, const float* __restrict__ bb2,
    float* __restrict__ h, int b0)
{
    __shared__ float xs[74];
    __shared__ float h1[128 * 68];
    const int t = threadIdx.x;
    const int bloc = blockIdx.y;
    const int s0 = blockIdx.x * 64;
    const float bnscale = rsqrtf(1.f + 1e-5f);
    if (t < 74) {
        int gs = s0 - 5 + t;
        xs[t] = (gs >= 0 && gs < SL) ? x[(size_t)(b0 + bloc) * SL + gs] : 0.f;
    }
    __syncthreads();
    for (int idx = t; idx < 128 * 68; idx += 256) {
        int c = idx / 68, ss = idx - c * 68;       // position s0-2+ss
        const float* wp = w1 + c * 7;
        float acc = b1[c];
#pragma unroll
        for (int i = 0; i < 7; ++i) acc += xs[ss + i] * wp[i];
        h1[c * 68 + ss] = geluf(acc * (g1[c] * bnscale) + bb1[c]);
    }
    __syncthreads();
    const float lfreq = logf(10000.f) / 256.f;
    for (int idx = t; idx < 256 * 64; idx += 256) {
        int d = idx >> 6, so = idx & 63;
        const float* wr = w2 + (size_t)d * 128 * 5;
        float acc = b2[d];
        for (int c = 0; c < 128; ++c) {
            const float* hp = &h1[c * 68 + so];
            const float* wp = wr + c * 5;
            acc += hp[0] * wp[0] + hp[1] * wp[1] + hp[2] * wp[2] +
                   hp[3] * wp[3] + hp[4] * wp[4];
        }
        float v = geluf(acc * (g2[d] * bnscale) + bb2[d]);
        int s = s0 + so;
        float freq = expf(-(float)(2 * (d >> 1)) * lfreq);
        float ang = (float)s * freq;
        v += (d & 1) ? cosf(ang) : sinf(ang);
        h[((size_t)bloc * SL + s) * 256 + d] = v;
    }
}

// ---------------------------------------------------------------------------
// Decoder: conv(256->128,k5,p2)+BN+GELU, conv(128->1,k3,p1), sigmoid
// ---------------------------------------------------------------------------
__global__ __launch_bounds__(256) void decode_conv(
    const float* __restrict__ h,
    const float* __restrict__ w1, const float* __restrict__ b1,
    const float* __restrict__ g3, const float* __restrict__ bb3,
    const float* __restrict__ w2, const float* __restrict__ b2,
    float* __restrict__ out, int b0)
{
    __shared__ float hs[38 * 260];   // rows s0-3 .. s0+34, padded stride
    __shared__ float h1[128 * 34];   // positions s0-1 .. s0+32
    const int t = threadIdx.x;
    const int bloc = blockIdx.y;
    const int s0 = blockIdx.x * 32;
    const float bnscale = rsqrtf(1.f + 1e-5f);
    for (int idx = t; idx < 38 * 256; idx += 256) {
        int j = idx >> 8, c = idx & 255;
        int gs = s0 - 3 + j;
        float v = 0.f;
        if (gs >= 0 && gs < SL) v = h[((size_t)bloc * SL + gs) * 256 + c];
        hs[j * 260 + c] = v;
    }
    __syncthreads();
    for (int idx = t; idx < 128 * 34; idx += 256) {
        int c = idx / 34, ss = idx - c * 34;
        float acc = b1[c];
        const float* wr = w1 + (size_t)c * 256 * 5;
        for (int k = 0; k < 256; ++k) {
            const float* hp = &hs[ss * 260 + k];
            const float* wp = wr + k * 5;
            acc += hp[0] * wp[0] + hp[260] * wp[1] + hp[520] * wp[2] +
                   hp[780] * wp[3] + hp[1040] * wp[4];
        }
        h1[c * 34 + ss] = geluf(acc * (g3[c] * bnscale) + bb3[c]);
    }
    __syncthreads();
    if (t < 32) {
        int so = t, s = s0 + so;
        float acc = b2[0];
        for (int c = 0; c < 128; ++c) {
            const float* hp = &h1[c * 34 + so];
            const float* wp = w2 + c * 3;
            acc += hp[0] * wp[0] + hp[1] * wp[1] + hp[2] * wp[2];
        }
        out[(size_t)(b0 + bloc) * SL + s] = 1.f / (1.f + expf(-acc));
    }
}

// ---------------------------------------------------------------------------
// Host launcher
// ---------------------------------------------------------------------------
extern "C" void kernel_launch(void* const* d_in, const int* in_sizes, int n_in,
                              void* d_out, int out_size, void* d_ws, size_t ws_size,
                              hipStream_t stream)
{
    (void)in_sizes; (void)n_in; (void)out_size;
    const float* x    = (const float*)d_in[0];
    const float* ew1  = (const float*)d_in[1];
    const float* eb1  = (const float*)d_in[2];
    const float* g1   = (const float*)d_in[3];
    const float* bb1  = (const float*)d_in[4];
    const float* ew2  = (const float*)d_in[5];
    const float* eb2  = (const float*)d_in[6];
    const float* g2   = (const float*)d_in[7];
    const float* bb2  = (const float*)d_in[8];
    const float* wq   = (const float*)d_in[9];
    const float* bq   = (const float*)d_in[10];
    const float* wk   = (const float*)d_in[11];
    const float* bk   = (const float*)d_in[12];
    const float* wv   = (const float*)d_in[13];
    const float* bv   = (const float*)d_in[14];
    const float* wo   = (const float*)d_in[15];
    const float* bo   = (const float*)d_in[16];
    const float* rf   = (const float*)d_in[17];
    const float* ln1g = (const float*)d_in[18];
    const float* ln1b = (const float*)d_in[19];
    const float* ln2g = (const float*)d_in[20];
    const float* ln2b = (const float*)d_in[21];
    const float* fw1  = (const float*)d_in[22];
    const float* fb1  = (const float*)d_in[23];
    const float* fw2  = (const float*)d_in[24];
    const float* fb2  = (const float*)d_in[25];
    const float* dw1  = (const float*)d_in[26];
    const float* db1  = (const float*)d_in[27];
    const float* g3   = (const float*)d_in[28];
    const float* bb3  = (const float*)d_in[29];
    const float* dw2  = (const float*)d_in[30];
    const float* db2  = (const float*)d_in[31];

    char* wsb = (char*)d_ws;
    size_t off = 0;
    auto alloc = [&](size_t bytes) -> size_t {
        size_t o = off;
        off = (off + bytes + 255) & ~(size_t)255;
        return o;
    };
    // Fixed (chunk-independent) region: packed bf16 weights
    size_t o_wqkvT = alloc((size_t)NL * 768 * 256 * 2);
    size_t o_woT   = alloc((size_t)NL * 256 * 256 * 2);
    size_t o_ff1T  = alloc((size_t)NL * 1024 * 256 * 2);
    size_t o_ff2T  = alloc((size_t)NL * 256 * 1024 * 2);
    size_t o_rfT   = alloc((size_t)NL * 8 * 64 * 32 * 2);
    size_t o_bqkv  = alloc((size_t)NL * 768 * 4);
    size_t fixedEnd = off;

    // Pick largest batch chunk that fits ws (model is per-sample independent)
    int bch = 1;
    const int cands[6] = {32, 16, 8, 4, 2, 1};
    for (int i = 0; i < 6; ++i) {
        int c = cands[i];
        size_t Mc = (size_t)c * SL;
        size_t o2 = fixedEnd;
        auto al2 = [&](size_t bytes) {
            o2 = (o2 + bytes + 255) & ~(size_t)255;
        };
        al2(Mc * 256 * 4);              // h
        al2(Mc * 256 * 2);              // xln
        al2(Mc * 768 * 2);              // qkv (attn aliases)
        al2(Mc * 512 * 2);              // qp  (mid aliases qp+kp)
        al2(Mc * 512 * 2);              // kp
        al2((size_t)c * 8 * 8 * 2048 * 4);  // pkv
        al2((size_t)c * 8 * 8 * 64 * 4);    // pks
        if (o2 <= ws_size) { bch = c; break; }
    }
    size_t Mc = (size_t)bch * SL;
    off = fixedEnd;
    size_t o_h   = alloc(Mc * 256 * 4);
    size_t o_xln = alloc(Mc * 256 * 2);
    size_t o_qkv = alloc(Mc * 768 * 2);
    size_t o_qp  = alloc(Mc * 512 * 2);
    size_t o_kp  = alloc(Mc * 512 * 2);
    size_t o_pkv = alloc((size_t)bch * 8 * 8 * 2048 * 4);
    size_t o_pks = alloc((size_t)bch * 8 * 8 * 64 * 4);

    bf16*  wqkvT = (bf16*)(wsb + o_wqkvT);
    bf16*  woT   = (bf16*)(wsb + o_woT);
    bf16*  ff1T  = (bf16*)(wsb + o_ff1T);
    bf16*  ff2T  = (bf16*)(wsb + o_ff2T);
    bf16*  rfT   = (bf16*)(wsb + o_rfT);
    float* bqkvF = (float*)(wsb + o_bqkv);
    float* hbuf  = (float*)(wsb + o_h);
    bf16*  xln   = (bf16*)(wsb + o_xln);
    bf16*  qkv   = (bf16*)(wsb + o_qkv);
    bf16*  qp    = (bf16*)(wsb + o_qp);
    bf16*  kp    = (bf16*)(wsb + o_kp);
    float* pkv   = (float*)(wsb + o_pkv);
    float* pks   = (float*)(wsb + o_pks);
    bf16*  attn  = qkv;   // alias: v consumed before attn written
    bf16*  mid   = qp;    // alias: qp/kp dead before ffn mid written

    // Pack weights (bf16, [N][K])
    for (int l = 0; l < NL; ++l) {
        pack_t<<<256, 256, 0, stream>>>(wq + (size_t)l * 65536, wqkvT + (size_t)l * 196608, 256, 256);
        pack_t<<<256, 256, 0, stream>>>(wk + (size_t)l * 65536, wqkvT + (size_t)l * 196608 + 65536, 256, 256);
        pack_t<<<256, 256, 0, stream>>>(wv + (size_t)l * 65536, wqkvT + (size_t)l * 196608 + 131072, 256, 256);
        pack_t<<<256, 256, 0, stream>>>(wo + (size_t)l * 65536, woT + (size_t)l * 65536, 256, 256);
        pack_t<<<1024, 256, 0, stream>>>(fw1 + (size_t)l * 262144, ff1T + (size_t)l * 262144, 256, 1024);
        pack_t<<<1024, 256, 0, stream>>>(fw2 + (size_t)l * 262144, ff2T + (size_t)l * 262144, 1024, 256);
        for (int hh = 0; hh < 8; ++hh)
            pack_t<<<8, 256, 0, stream>>>(rf + (size_t)(l * 8 + hh) * 2048,
                                          rfT + (size_t)(l * 8 + hh) * 2048, 32, 64);
    }
    pack_bias<<<12, 256, 0, stream>>>(bq, bk, bv, bqkvF);

    const dim3 blk(256);
    for (int b0 = 0; b0 < 32; b0 += bch) {
        embed_conv<<<dim3(SL / 64, bch), blk, 0, stream>>>(
            x, ew1, eb1, g1, bb1, ew2, eb2, g2, bb2, hbuf, b0);
        for (int l = 0; l < NL; ++l) {
            ln_rows<<<dim3((unsigned)(Mc / 8)), blk, 0, stream>>>(
                hbuf, ln1g + l * 256, ln1b + l * 256, xln, (int)Mc);
            // QKV fused GEMM: [Mc,256] x [256,768]
            gemm_wmma<0, 128, 4><<<dim3(6, (unsigned)(Mc / 128), 1), blk, 0, stream>>>(
                xln, 256, wqkvT + (size_t)l * 196608, bqkvF + l * 768,
                qkv, 768, (int)Mc, 768, 256, 1, 0, 0, 0, 0, 0, 0);
            // qp = softplus(q @ rf), batched over (b,h), K=32 (one WMMA step)
            gemm_wmma<3, 64, 8><<<dim3(1, 32, bch * 8), blk, 0, stream>>>(
                qkv, 768, rfT + (size_t)l * 16384, nullptr,
                qp, 64, SL, 64, 32,
                8, (size_t)SL * 768, 32, 0, 2048,
                (size_t)8 * SL * 64, (size_t)SL * 64);
            gemm_wmma<3, 64, 8><<<dim3(1, 32, bch * 8), blk, 0, stream>>>(
                qkv + 256, 768, rfT + (size_t)l * 16384, nullptr,
                kp, 64, SL, 64, 32,
                8, (size_t)SL * 768, 32, 0, 2048,
                (size_t)8 * SL * 64, (size_t)SL * 64);
            kv_partial<<<dim3(8, bch * 8), blk, 0, stream>>>(qkv, kp, pkv, pks);
            attn_combine<<<dim3(SL / 256, bch * 8), blk, 0, stream>>>(qp, pkv, pks, attn);
            // h += attn @ wo + bo
            gemm_wmma<2, 128, 4><<<dim3(2, (unsigned)(Mc / 128), 1), blk, 0, stream>>>(
                attn, 256, woT + (size_t)l * 65536, bo + l * 256,
                hbuf, 256, (int)Mc, 256, 256, 1, 0, 0, 0, 0, 0, 0);
            // FFN
            ln_rows<<<dim3((unsigned)(Mc / 8)), blk, 0, stream>>>(
                hbuf, ln2g + l * 256, ln2b + l * 256, xln, (int)Mc);
            gemm_wmma<1, 128, 4><<<dim3(8, (unsigned)(Mc / 128), 1), blk, 0, stream>>>(
                xln, 256, ff1T + (size_t)l * 262144, fb1 + l * 1024,
                mid, 1024, (int)Mc, 1024, 256, 1, 0, 0, 0, 0, 0, 0);
            gemm_wmma<2, 128, 4><<<dim3(2, (unsigned)(Mc / 128), 1), blk, 0, stream>>>(
                mid, 1024, ff2T + (size_t)l * 262144, fb2 + l * 256,
                hbuf, 256, (int)Mc, 256, 1024, 1, 0, 0, 0, 0, 0, 0);
        }
        decode_conv<<<dim3(SL / 32, bch), blk, 0, stream>>>(
            hbuf, dw1, db1, g3, bb3, dw2, db2, (float*)d_out, b0);
    }
}
